// LongformerPairClassifier_86586540688061
// MI455X (gfx1250) — matrix operand order repaired and checked by reading, not compiled, not run on hardware
//
#include <hip/hip_runtime.h>
#include <cstdint>
#include <cstddef>

// ---------------------------------------------------------------------------
// Types & helpers
// ---------------------------------------------------------------------------
typedef unsigned short u16;
typedef __attribute__((ext_vector_type(16))) __bf16 v16bf;
typedef __attribute__((ext_vector_type(8)))  float  v8f;

union AFrag { v16bf v; u16 u[16]; uint4 q[2]; };

#define NHEADS 12
#define GTOK   64
#define NEGBIG (-1.0e9f)

__device__ inline u16 f2bf(float f) {
  union { float f; uint32_t u; } x; x.f = f;
  uint32_t u = x.u;
  uint32_t r = u + 0x7FFFu + ((u >> 16) & 1u);
  if ((u & 0x7F800000u) == 0x7F800000u) r = u;   // inf/nan passthrough
  return (u16)(r >> 16);
}

// A-matrix (16x32 bf16) K mapping: lanes 0-15 hold K {0..7,16..23}, lanes 16-31 hold K {8..15,24..31}
__device__ inline int kmapA(int j, int lhalf) { return ((j < 8) ? j : j + 8) + lhalf * 8; }

__device__ inline v8f wmma_bf16(const AFrag& a, const AFrag& b, v8f c) {
  return __builtin_amdgcn_wmma_f32_16x16x32_bf16(false, a.v, false, b.v, (short)0, c, false, false);
}

__device__ inline float bfly_max16(float v) {
#pragma unroll
  for (int m = 1; m < 16; m <<= 1) v = fmaxf(v, __shfl_xor(v, m, 32));
  return v;
}
__device__ inline float bfly_add16(float v) {
#pragma unroll
  for (int m = 1; m < 16; m <<= 1) v += __shfl_xor(v, m, 32);
  return v;
}

__device__ inline float gelu_f(float x) {
  float x3 = x * x * x;
  return 0.5f * x * (1.0f + tanhf(0.7978845608028654f * (x + 0.044715f * x3)));
}

__device__ inline float block_reduce_sum_256(float v, float* red) {
  int lane = threadIdx.x & 31, wid = threadIdx.x >> 5;
#pragma unroll
  for (int m = 16; m > 0; m >>= 1) v += __shfl_xor(v, m, 32);
  if (lane == 0) red[wid] = v;
  __syncthreads();
  float tot = 0.0f;
#pragma unroll
  for (int i = 0; i < 8; ++i) tot += red[i];
  __syncthreads();
  return tot;
}

__device__ inline void and_mask4(uint4& v, unsigned m) {
  v.x &= m; v.y &= m; v.z &= m; v.w &= m;
}

// ---------------------------------------------------------------------------
// f32 -> bf16 transpose-convert:  out[N][K] bf16 = in[K][N] f32 transposed
// Tiled 32x32 through LDS for coalescing on both sides.  K,N multiples of 32.
// ---------------------------------------------------------------------------
__global__ __launch_bounds__(256) void convert_t_kernel(const float* __restrict__ in,
                                                        u16* __restrict__ out,
                                                        int K, int N) {
  __shared__ float tile[32][33];
  int kb = blockIdx.x * 32;
  int nb = blockIdx.y * 32;
  int tx = threadIdx.x & 31;
  int ty = threadIdx.x >> 5;
#pragma unroll
  for (int i = 0; i < 32; i += 8)
    tile[ty + i][tx] = in[(size_t)(kb + ty + i) * N + nb + tx];
  __syncthreads();
#pragma unroll
  for (int i = 0; i < 32; i += 8)
    out[(size_t)(nb + ty + i) * K + kb + tx] = f2bf(tile[tx][ty + i]);
}

// ---------------------------------------------------------------------------
// Embedding + LayerNorm  (one block per token, 256 threads, h=768)
// ---------------------------------------------------------------------------
__global__ __launch_bounds__(256) void embed_ln_kernel(const int* __restrict__ ids,
                                                       const float* __restrict__ wemb,
                                                       const float* __restrict__ pemb,
                                                       const float* __restrict__ g,
                                                       const float* __restrict__ bta,
                                                       float* __restrict__ X,
                                                       u16* __restrict__ Xb, int S) {
  const int h = 768;
  int tok = blockIdx.x;
  int s = tok % S;
  int id = ids[tok];
  __shared__ float red[8];
  float vals[3];
  float sum = 0.0f;
#pragma unroll
  for (int i = 0; i < 3; ++i) {
    int c = threadIdx.x + i * 256;
    float v = wemb[(size_t)id * h + c] + pemb[(size_t)s * h + c];
    vals[i] = v; sum += v;
  }
  float mean = block_reduce_sum_256(sum, red) * (1.0f / h);
  float var = 0.0f;
#pragma unroll
  for (int i = 0; i < 3; ++i) { float d = vals[i] - mean; var += d * d; }
  var = block_reduce_sum_256(var, red) * (1.0f / h);
  float rstd = rsqrtf(var + 1e-5f);
#pragma unroll
  for (int i = 0; i < 3; ++i) {
    int c = threadIdx.x + i * 256;
    float o = (vals[i] - mean) * rstd * g[c] + bta[c];
    X[(size_t)tok * h + c] = o;
    Xb[(size_t)tok * h + c] = f2bf(o);
  }
}

// ---------------------------------------------------------------------------
// Residual add + LayerNorm (in place on X), refresh bf16 copy
// ---------------------------------------------------------------------------
__global__ __launch_bounds__(256) void add_ln_kernel(float* __restrict__ X,
                                                     const float* __restrict__ Y,
                                                     const float* __restrict__ g,
                                                     const float* __restrict__ bta,
                                                     u16* __restrict__ Xb) {
  const int h = 768;
  int tok = blockIdx.x;
  __shared__ float red[8];
  float vals[3];
  float sum = 0.0f;
#pragma unroll
  for (int i = 0; i < 3; ++i) {
    int c = threadIdx.x + i * 256;
    float v = X[(size_t)tok * h + c] + Y[(size_t)tok * h + c];
    vals[i] = v; sum += v;
  }
  float mean = block_reduce_sum_256(sum, red) * (1.0f / h);
  float var = 0.0f;
#pragma unroll
  for (int i = 0; i < 3; ++i) { float d = vals[i] - mean; var += d * d; }
  var = block_reduce_sum_256(var, red) * (1.0f / h);
  float rstd = rsqrtf(var + 1e-5f);
#pragma unroll
  for (int i = 0; i < 3; ++i) {
    int c = threadIdx.x + i * 256;
    float o = (vals[i] - mean) * rstd * g[c] + bta[c];
    X[(size_t)tok * h + c] = o;
    Xb[(size_t)tok * h + c] = f2bf(o);
  }
}

// ---------------------------------------------------------------------------
// Generic WMMA GEMM:  out = act((A[MxK] @ Wt[NxK]^T + bias) * scale)
// A bf16 row-major [M][K]; Wt bf16 TRANSPOSED row-major [N][K].
// 256 threads = 8 waves (2x4), block tile 64x128, wave tile 32x32, BK=32.
// All LDS fragment reads are contiguous 16B (ds_load_b128).
// ---------------------------------------------------------------------------
__global__ __launch_bounds__(256) void gemm_kernel(
    const u16* __restrict__ A, const u16* __restrict__ Wt, const float* __restrict__ bias,
    float* __restrict__ outF, u16* __restrict__ outB,
    int M, int N, int K, int act, float scale, int headScatter, int S) {
  __shared__ __align__(16) u16 As[64][48];    // 6 KB (96B rows, 16B aligned)
  __shared__ __align__(16) u16 Bt[128][48];   // 12 KB
  const int bm = blockIdx.y * 64;
  const int bn = blockIdx.x * 128;
  const int wave = threadIdx.x >> 5, lane = threadIdx.x & 31;
  const int wm = (wave >> 2) * 32, wn = (wave & 3) * 32;
  const int lhalf = lane >> 4, lrow = lane & 15;
  v8f acc[2][2];
#pragma unroll
  for (int i = 0; i < 2; ++i)
#pragma unroll
    for (int j = 0; j < 2; ++j) { v8f z = {0,0,0,0,0,0,0,0}; acc[i][j] = z; }

  for (int k0 = 0; k0 < K; k0 += 32) {
    __syncthreads();
    {
      // A tile 64x32: one uint4 (8 bf16) per thread, branchless edge zeroing
      int r = threadIdx.x >> 2;
      int cc = (threadIdx.x & 3) * 8;
      int rr = bm + r;
      bool okrow = rr < M;
      int rc = okrow ? rr : 0;
      const u16* srcA = A + (size_t)rc * K + k0 + cc;
      uint4 av = *(const uint4*)srcA;
      and_mask4(av, okrow ? 0xFFFFFFFFu : 0u);
      *(uint4*)&As[r][cc] = av;
      if (k0 + 32 < K) __builtin_prefetch(srcA + 32, 0, 1);
    }
    {
      // B tile 128x32 from transposed weights: two uint4 per thread
#pragma unroll
      for (int i = 0; i < 2; ++i) {
        int idx = threadIdx.x * 2 + i;
        int n = idx >> 2;
        int cc = (idx & 3) * 8;
        const u16* srcB = Wt + (size_t)(bn + n) * K + k0 + cc;
        *(uint4*)&Bt[n][cc] = *(const uint4*)srcB;
        if (k0 + 32 < K) __builtin_prefetch(srcB + 32, 0, 1);
      }
    }
    __syncthreads();
    AFrag af[2], bf[2];
#pragma unroll
    for (int mt = 0; mt < 2; ++mt) {
      const u16* row = &As[wm + mt * 16 + lrow][0];
      af[mt].q[0] = *(const uint4*)(row + lhalf * 8);
      af[mt].q[1] = *(const uint4*)(row + 16 + lhalf * 8);
    }
#pragma unroll
    for (int nt = 0; nt < 2; ++nt) {
      const u16* row = &Bt[wn + nt * 16 + lrow][0];
      bf[nt].q[0] = *(const uint4*)(row + lhalf * 8);
      bf[nt].q[1] = *(const uint4*)(row + 16 + lhalf * 8);
    }
#pragma unroll
    for (int mt = 0; mt < 2; ++mt)
#pragma unroll
      for (int nt = 0; nt < 2; ++nt) acc[mt][nt] = wmma_bf16(af[mt], bf[nt], acc[mt][nt]);
  }

#pragma unroll
  for (int mt = 0; mt < 2; ++mt)
#pragma unroll
    for (int nt = 0; nt < 2; ++nt)
#pragma unroll
      for (int r = 0; r < 8; ++r) {
        int gm = bm + wm + mt * 16 + r + lhalf * 8;
        int gn = bn + wn + nt * 16 + lrow;
        if (gm >= M) continue;
        float v = (acc[mt][nt][r] + bias[gn]) * scale;
        if (act == 1) v = gelu_f(v);
        else if (act == 2) v = fmaxf(v, 0.0f);
        if (outF) outF[(size_t)gm * N + gn] = v;
        if (outB) {
          if (headScatter) {
            int b = gm / S, s2 = gm % S;
            int hh = gn >> 6, dc = gn & 63;
            outB[(((size_t)(b * NHEADS + hh) * S + s2) << 6) + dc] = f2bf(v);
          } else {
            outB[(size_t)gm * N + gn] = f2bf(v);
          }
        }
      }
}

// ---------------------------------------------------------------------------
// Fused band + global-key attention (flash style, online softmax).
// Grid: (chunk n=16, b*h=24).  256 threads = 8 waves, each wave = 32 query rows.
// Key stream = 2 blocks of 32 global keys (k[:, :G]) + 24 blocks of 32 band keys.
// K staged row-major (scores B-frag contiguous); V staged transposed (PV B-frag contiguous).
// ---------------------------------------------------------------------------
__global__ __launch_bounds__(256) void attn_band_kernel(
    const u16* __restrict__ Q, const u16* __restrict__ K, const u16* __restrict__ V,
    const int* __restrict__ am, float* __restrict__ O, int S) {
  const int ni = blockIdx.x;           // chunk
  const int bh = blockIdx.y;
  const int b = bh / NHEADS;
  const u16* Qh = Q + (size_t)bh * S * 64;
  const u16* Kh = K + (size_t)bh * S * 64;
  const u16* Vh = V + (size_t)bh * S * 64;

  __shared__ __align__(16) u16 qs[256 * 64];    // 32 KB
  __shared__ __align__(16) u16 kbuf[32 * 64];   // 4 KB   [key][d]
  __shared__ __align__(16) u16 vt[64 * 48];     // 6 KB   [d][key] (stride 48)
  __shared__ int kval[32];
  __shared__ __align__(16) u16 pstage[8][16 * 32];  // 8 KB per-wave prob staging

  {
    const uint4* qsrc = (const uint4*)(Qh + (size_t)(ni * 256) * 64);
    uint4* qdst = (uint4*)qs;
    for (int i = threadIdx.x; i < 2048; i += 256) qdst[i] = qsrc[i];
  }
  __syncthreads();

  const int wave = threadIdx.x >> 5, lane = threadIdx.x & 31;
  const int lhalf = lane >> 4, lrow = lane & 15;

  AFrag aq[2][2];
#pragma unroll
  for (int mt = 0; mt < 2; ++mt)
#pragma unroll
    for (int ks = 0; ks < 2; ++ks) {
      const u16* row = qs + (size_t)(wave * 32 + mt * 16 + lrow) * 64 + ks * 32;
      aq[mt][ks].q[0] = *(const uint4*)(row + lhalf * 8);
      aq[mt][ks].q[1] = *(const uint4*)(row + 16 + lhalf * 8);
    }

  v8f o[2][4];
  float rm[2][8], rl[2][8];
#pragma unroll
  for (int mt = 0; mt < 2; ++mt) {
#pragma unroll
    for (int ot = 0; ot < 4; ++ot) { v8f z = {0,0,0,0,0,0,0,0}; o[mt][ot] = z; }
#pragma unroll
    for (int r = 0; r < 8; ++r) { rm[mt][r] = -3.0e38f; rl[mt][r] = 0.0f; }
  }

  const int NBLK = 2 + 24;  // 64 global + 768 band keys, blocks of 32
  for (int kb = 0; kb < NBLK; ++kb) {
    __syncthreads();
    {
      // stage 32 keys: branchless b128 loads with AND-mask zeroing
      int kr = threadIdx.x >> 3;          // key row 0..31
      int cc = (threadIdx.x & 7) * 8;     // d offset 0..56
      int p; bool inr;
      if (kb < 2) { p = kb * 32 + kr; inr = true; }
      else { int j = (kb - 2) * 32 + kr; p = ni * 256 + j - 256; inr = (p >= 0 && p < S); }
      int pc = inr ? p : 0;
      unsigned msk = inr ? 0xFFFFFFFFu : 0u;
      uint4 kv = *(const uint4*)(Kh + (size_t)pc * 64 + cc);
      uint4 vv = *(const uint4*)(Vh + (size_t)pc * 64 + cc);
      and_mask4(kv, msk);
      and_mask4(vv, msk);
      *(uint4*)&kbuf[kr * 64 + cc] = kv;
      const u16* vvp = (const u16*)&vv;
#pragma unroll
      for (int j = 0; j < 8; ++j) vt[(cc + j) * 48 + kr] = vvp[j];
      if (threadIdx.x < 32) {
        int t = threadIdx.x; int ok;
        if (kb < 2) { int pp = kb * 32 + t; ok = (am[(size_t)b * S + pp] > 0) ? 1 : 0; }
        else {
          int j = (kb - 2) * 32 + t; int pp = ni * 256 + j - 256;
          ok = (pp >= 0 && pp < S && pp >= GTOK && am[(size_t)b * S + pp] > 0) ? 1 : 0;
        }
        kval[t] = ok;
      }
    }
    __syncthreads();

    v8f sc[2][2];
#pragma unroll
    for (int mt = 0; mt < 2; ++mt)
#pragma unroll
      for (int nt = 0; nt < 2; ++nt) { v8f z = {0,0,0,0,0,0,0,0}; sc[mt][nt] = z; }
#pragma unroll
    for (int nt = 0; nt < 2; ++nt)
#pragma unroll
      for (int ks = 0; ks < 2; ++ks) {
        AFrag bk;
        const u16* row = kbuf + (size_t)(nt * 16 + lrow) * 64 + ks * 32;
        bk.q[0] = *(const uint4*)(row + lhalf * 8);
        bk.q[1] = *(const uint4*)(row + 16 + lhalf * 8);
#pragma unroll
        for (int mt = 0; mt < 2; ++mt) sc[mt][nt] = wmma_bf16(aq[mt][ks], bk, sc[mt][nt]);
      }

    // mask
#pragma unroll
    for (int mt = 0; mt < 2; ++mt)
#pragma unroll
      for (int nt = 0; nt < 2; ++nt)
#pragma unroll
        for (int r = 0; r < 8; ++r) {
          int ci = wave * 32 + mt * 16 + r + lhalf * 8;
          int kk = nt * 16 + lrow;
          bool ok = kval[kk] != 0;
          if (kb >= 2) {
            int j = (kb - 2) * 32 + kk;
            int rel = j - 256 - ci;
            ok = ok && (rel >= -256) && (rel <= 256);
          }
          if (!ok) sc[mt][nt][r] = NEGBIG;
        }

    // online softmax + P@V
#pragma unroll
    for (int mt = 0; mt < 2; ++mt) {
      float p0[8], p1[8];
#pragma unroll
      for (int r = 0; r < 8; ++r) {
        float bmv = fmaxf(sc[mt][0][r], sc[mt][1][r]);
        bmv = bfly_max16(bmv);
        float nm = fmaxf(rm[mt][r], bmv);
        float alpha = __expf(rm[mt][r] - nm);
        float q0 = __expf(sc[mt][0][r] - nm);
        float q1 = __expf(sc[mt][1][r] - nm);
        float rs = bfly_add16(q0 + q1);
        rl[mt][r] = rl[mt][r] * alpha + rs;
        rm[mt][r] = nm;
#pragma unroll
        for (int ot = 0; ot < 4; ++ot) o[mt][ot][r] *= alpha;
        p0[r] = q0; p1[r] = q1;
      }
      u16* ps = &pstage[wave][0];
#pragma unroll
      for (int r = 0; r < 8; ++r) {
        ps[(r + lhalf * 8) * 32 + lrow] = f2bf(p0[r]);
        ps[(r + lhalf * 8) * 32 + 16 + lrow] = f2bf(p1[r]);
      }
      asm volatile("s_wait_dscnt 0x0" ::: "memory");
      AFrag pa;
      {
        const u16* row = ps + lrow * 32;
        pa.q[0] = *(const uint4*)(row + lhalf * 8);
        pa.q[1] = *(const uint4*)(row + 16 + lhalf * 8);
      }
#pragma unroll
      for (int ot = 0; ot < 4; ++ot) {
        AFrag bv;
        const u16* row = vt + (size_t)(ot * 16 + lrow) * 48;
        bv.q[0] = *(const uint4*)(row + lhalf * 8);
        bv.q[1] = *(const uint4*)(row + 16 + lhalf * 8);
        o[mt][ot] = wmma_bf16(pa, bv, o[mt][ot]);
      }
    }
  }

#pragma unroll
  for (int mt = 0; mt < 2; ++mt)
#pragma unroll
    for (int ot = 0; ot < 4; ++ot)
#pragma unroll
      for (int r = 0; r < 8; ++r) {
        int row = ni * 256 + wave * 32 + mt * 16 + r + lhalf * 8;
        O[((size_t)bh * S + row) * 64 + ot * 16 + lrow] = o[mt][ot][r] / rl[mt][r];
      }
}

// ---------------------------------------------------------------------------
// Global-token attention (G=64 query rows attend to all S keys), overwrites
// O rows [0,G).  Grid: b*h.  128 threads = 4 waves, each wave = 16 rows.
// ---------------------------------------------------------------------------
__global__ __launch_bounds__(128) void attn_global_kernel(
    const u16* __restrict__ Qg, const u16* __restrict__ Kg, const u16* __restrict__ Vg,
    const int* __restrict__ am, float* __restrict__ O, int S) {
  const int bh = blockIdx.x;
  const int b = bh / NHEADS;
  const u16* Qh = Qg + (size_t)bh * S * 64;
  const u16* Kh = Kg + (size_t)bh * S * 64;
  const u16* Vh = Vg + (size_t)bh * S * 64;

  __shared__ __align__(16) u16 qs[64 * 64];     // 8 KB
  __shared__ __align__(16) u16 kbuf[32 * 64];   // 4 KB
  __shared__ __align__(16) u16 vt[64 * 48];     // 6 KB
  __shared__ int kval[32];
  __shared__ __align__(16) u16 pstage[4][16 * 32];

  {
    const uint4* qsrc = (const uint4*)Qh;
    uint4* qdst = (uint4*)qs;
    for (int i = threadIdx.x; i < 512; i += 128) qdst[i] = qsrc[i];
  }
  __syncthreads();

  const int wave = threadIdx.x >> 5, lane = threadIdx.x & 31;
  const int lhalf = lane >> 4, lrow = lane & 15;

  AFrag aq[2];
#pragma unroll
  for (int ks = 0; ks < 2; ++ks) {
    const u16* row = qs + (size_t)(wave * 16 + lrow) * 64 + ks * 32;
    aq[ks].q[0] = *(const uint4*)(row + lhalf * 8);
    aq[ks].q[1] = *(const uint4*)(row + 16 + lhalf * 8);
  }

  v8f o[4];
  float rm[8], rl[8];
#pragma unroll
  for (int ot = 0; ot < 4; ++ot) { v8f z = {0,0,0,0,0,0,0,0}; o[ot] = z; }
#pragma unroll
  for (int r = 0; r < 8; ++r) { rm[r] = -3.0e38f; rl[r] = 0.0f; }

  const int NBLK = S / 32;
  for (int kb = 0; kb < NBLK; ++kb) {
    __syncthreads();
    {
#pragma unroll
      for (int i = 0; i < 2; ++i) {
        int idx = threadIdx.x * 2 + i;      // 0..255
        int kr = idx >> 3;
        int cc = (idx & 7) * 8;
        int p = kb * 32 + kr;
        uint4 kv = *(const uint4*)(Kh + (size_t)p * 64 + cc);
        uint4 vv = *(const uint4*)(Vh + (size_t)p * 64 + cc);
        *(uint4*)&kbuf[kr * 64 + cc] = kv;
        const u16* vvp = (const u16*)&vv;
#pragma unroll
        for (int j = 0; j < 8; ++j) vt[(cc + j) * 48 + kr] = vvp[j];
      }
      if (threadIdx.x < 32)
        kval[threadIdx.x] = (am[(size_t)b * S + kb * 32 + threadIdx.x] > 0) ? 1 : 0;
    }
    __syncthreads();

    v8f sc[2];
#pragma unroll
    for (int nt = 0; nt < 2; ++nt) { v8f z = {0,0,0,0,0,0,0,0}; sc[nt] = z; }
#pragma unroll
    for (int nt = 0; nt < 2; ++nt)
#pragma unroll
      for (int ks = 0; ks < 2; ++ks) {
        AFrag bk;
        const u16* row = kbuf + (size_t)(nt * 16 + lrow) * 64 + ks * 32;
        bk.q[0] = *(const uint4*)(row + lhalf * 8);
        bk.q[1] = *(const uint4*)(row + 16 + lhalf * 8);
        sc[nt] = wmma_bf16(aq[ks], bk, sc[nt]);
      }
#pragma unroll
    for (int nt = 0; nt < 2; ++nt)
#pragma unroll
      for (int r = 0; r < 8; ++r)
        if (!kval[nt * 16 + lrow]) sc[nt][r] = NEGBIG;

    float p0[8], p1[8];
#pragma unroll
    for (int r = 0; r < 8; ++r) {
      float bmv = fmaxf(sc[0][r], sc[1][r]);
      bmv = bfly_max16(bmv);
      float nm = fmaxf(rm[r], bmv);
      float alpha = __expf(rm[r] - nm);
      float q0 = __expf(sc[0][r] - nm);
      float q1 = __expf(sc[1][r] - nm);
      float rs = bfly_add16(q0 + q1);
      rl[r] = rl[r] * alpha + rs;
      rm[r] = nm;
#pragma unroll
      for (int ot = 0; ot < 4; ++ot) o[ot][r] *= alpha;
      p0[r] = q0; p1[r] = q1;
    }
    u16* ps = &pstage[wave][0];
#pragma unroll
    for (int r = 0; r < 8; ++r) {
      ps[(r + lhalf * 8) * 32 + lrow] = f2bf(p0[r]);
      ps[(r + lhalf * 8) * 32 + 16 + lrow] = f2bf(p1[r]);
    }
    asm volatile("s_wait_dscnt 0x0" ::: "memory");
    AFrag pa;
    {
      const u16* row = ps + lrow * 32;
      pa.q[0] = *(const uint4*)(row + lhalf * 8);
      pa.q[1] = *(const uint4*)(row + 16 + lhalf * 8);
    }
#pragma unroll
    for (int ot = 0; ot < 4; ++ot) {
      AFrag bv;
      const u16* row = vt + (size_t)(ot * 16 + lrow) * 48;
      bv.q[0] = *(const uint4*)(row + lhalf * 8);
      bv.q[1] = *(const uint4*)(row + 16 + lhalf * 8);
      o[ot] = wmma_bf16(pa, bv, o[ot]);
    }
  }

#pragma unroll
  for (int ot = 0; ot < 4; ++ot)
#pragma unroll
    for (int r = 0; r < 8; ++r) {
      int s2 = wave * 16 + r + lhalf * 8;     // < 64
      O[((size_t)bh * S + s2) * 64 + ot * 16 + lrow] = o[ot][r] / rl[r];
    }
}

// ---------------------------------------------------------------------------
// Merge heads: O [B][H][S][64] f32  ->  Ob [B*S][768] bf16
// ---------------------------------------------------------------------------
__global__ __launch_bounds__(256) void merge_heads_kernel(const float* __restrict__ O,
                                                          u16* __restrict__ Ob, int S) {
  size_t i = (size_t)blockIdx.x * 256 + threadIdx.x;
  int col = (int)(i % 768);
  size_t tok = i / 768;
  int b = (int)(tok / S), s = (int)(tok % S);
  int hh = col >> 6, dc = col & 63;
  Ob[i] = f2bf(O[((size_t)(b * NHEADS + hh) * S + s) * 64 + dc]);
}

// ---------------------------------------------------------------------------
// Gather pair embeddings: PE[b*P+p] = concat(X[b,i0], X[b,i1]) as bf16
// ---------------------------------------------------------------------------
__global__ __launch_bounds__(256) void pair_gather_kernel(const float* __restrict__ X,
                                                          const int* __restrict__ pidx,
                                                          u16* __restrict__ PE, int S) {
  const int h = 768, P = 128;
  int pr = blockIdx.x;                 // 0..255
  int b = pr / P;
  int i0 = pidx[pr * 2 + 0];
  int i1 = pidx[pr * 2 + 1];
  for (int c = threadIdx.x; c < 2 * h; c += 256) {
    float v = (c < h) ? X[((size_t)b * S + i0) * h + c]
                      : X[((size_t)b * S + i1) * h + (c - h)];
    PE[(size_t)pr * 2 * h + c] = f2bf(v);
  }
}

// ---------------------------------------------------------------------------
// Final tiny GEMM: out[256,3] = Z[256,768] @ F4[768,3] + f4
// ---------------------------------------------------------------------------
__global__ __launch_bounds__(256) void final_kernel(const float* __restrict__ Z,
                                                    const float* __restrict__ F4,
                                                    const float* __restrict__ f4,
                                                    float* __restrict__ out) {
  int idx = blockIdx.x * 256 + threadIdx.x;
  if (idx >= 256 * 3) return;
  int m = idx / 3, j = idx % 3;
  float acc = f4[j];
  for (int k = 0; k < 768; ++k) acc += Z[(size_t)m * 768 + k] * F4[(size_t)k * 3 + j];
  out[idx] = acc;
}

// ---------------------------------------------------------------------------
// Host orchestration
// ---------------------------------------------------------------------------
extern "C" void kernel_launch(void* const* d_in, const int* in_sizes, int n_in,
                              void* d_out, int out_size, void* d_ws, size_t ws_size,
                              hipStream_t stream) {
  (void)in_sizes; (void)n_in; (void)out_size; (void)ws_size;
  const int B = 2, S = 4096, h = 768, dff = 3072, L = 4, P = 128;
  const int BS = B * S;

  const int*   ids  = (const int*)d_in[0];
  const int*   am   = (const int*)d_in[1];
  const int*   pidx = (const int*)d_in[3];
  const float* wemb = (const float*)d_in[6];
  const float* pemb = (const float*)d_in[7];
  const float* lnEg = (const float*)d_in[8];
  const float* lnEb = (const float*)d_in[9];
  const float* Wq  = (const float*)d_in[10]; const float* bq  = (const float*)d_in[11];
  const float* Wk  = (const float*)d_in[12]; const float* bk  = (const float*)d_in[13];
  const float* Wv  = (const float*)d_in[14]; const float* bv  = (const float*)d_in[15];
  const float* Wo  = (const float*)d_in[16]; const float* bo  = (const float*)d_in[17];
  const float* Wqg = (const float*)d_in[18]; const float* bqg = (const float*)d_in[19];
  const float* Wkg = (const float*)d_in[20]; const float* bkg = (const float*)d_in[21];
  const float* Wvg = (const float*)d_in[22]; const float* bvg = (const float*)d_in[23];
  const float* ln1g = (const float*)d_in[24]; const float* ln1b = (const float*)d_in[25];
  const float* W1  = (const float*)d_in[26]; const float* b1  = (const float*)d_in[27];
  const float* W2  = (const float*)d_in[28]; const float* b2  = (const float*)d_in[29];
  const float* ln2g = (const float*)d_in[30]; const float* ln2b = (const float*)d_in[31];
  const float* F1  = (const float*)d_in[32]; const float* f1  = (const float*)d_in[33];
  const float* F2  = (const float*)d_in[34]; const float* f2  = (const float*)d_in[35];
  const float* F3  = (const float*)d_in[36]; const float* f3  = (const float*)d_in[37];
  const float* F4  = (const float*)d_in[38]; const float* f4  = (const float*)d_in[39];

  char* ws = (char*)d_ws;
  size_t off = 0;
  auto alloc = [&](size_t bytes) -> void* {
    void* p = ws + off;
    off += (bytes + 255) & ~(size_t)255;
    return p;
  };
  float* X   = (float*)alloc((size_t)BS * h * 4);
  u16*   Xb  = (u16*)  alloc((size_t)BS * h * 2);
  float* Y   = (float*)alloc((size_t)BS * dff * 4);
  u16*   Yb  = (u16*)  alloc((size_t)BS * dff * 2);
  u16*   Qb  = (u16*)  alloc((size_t)BS * h * 2);
  u16*   Kb  = (u16*)  alloc((size_t)BS * h * 2);
  u16*   Vb  = (u16*)  alloc((size_t)BS * h * 2);
  u16*   Qgb = (u16*)  alloc((size_t)BS * h * 2);
  u16*   Kgb = (u16*)  alloc((size_t)BS * h * 2);
  u16*   Vgb = (u16*)  alloc((size_t)BS * h * 2);
  float* O   = (float*)alloc((size_t)BS * h * 4);
  u16*   Ob  = (u16*)  alloc((size_t)BS * h * 2);
  u16*   Wb  = (u16*)  alloc((size_t)2304 * 1536 * 2);   // transposed bf16 weight scratch
  u16*   PEb = (u16*)  alloc((size_t)256 * 1536 * 2);
  u16*   Z1b = (u16*)  alloc((size_t)256 * 2304 * 2);
  u16*   Z2b = (u16*)  alloc((size_t)256 * 1536 * 2);
  float* Z3  = (float*)alloc((size_t)256 * 768 * 4);

  // transpose-convert weight [K2][N2] f32 -> Wb [N2][K2] bf16
  auto cvtT = [&](const float* src, int K2, int N2) {
    convert_t_kernel<<<dim3(K2 / 32, N2 / 32), dim3(256), 0, stream>>>(src, Wb, K2, N2);
  };
  auto gemm = [&](const u16* A, const float* bias, float* oF, u16* oB,
                  int M, int N, int K, int act, float scl, int hs) {
    gemm_kernel<<<dim3(N / 128, M / 64), dim3(256), 0, stream>>>(
        A, Wb, bias, oF, oB, M, N, K, act, scl, hs, S);
  };

  // embedding + LN
  embed_ln_kernel<<<dim3(BS), dim3(256), 0, stream>>>(ids, wemb, pemb, lnEg, lnEb, X, Xb, S);

  const float qscale = 0.125f;  // 1/sqrt(64)
  for (int l = 0; l < L; ++l) {
    size_t wof = (size_t)l * h * h, bof = (size_t)l * h;
    // QKV + global QKV projections (head-scattered bf16 outputs)
    cvtT(Wq + wof, h, h);  gemm(Xb, bq + bof, nullptr, Qb,  BS, h, h, 0, qscale, 1);
    cvtT(Wk + wof, h, h);  gemm(Xb, bk + bof, nullptr, Kb,  BS, h, h, 0, 1.0f,   1);
    cvtT(Wv + wof, h, h);  gemm(Xb, bv + bof, nullptr, Vb,  BS, h, h, 0, 1.0f,   1);
    cvtT(Wqg + wof, h, h); gemm(Xb, bqg + bof, nullptr, Qgb, BS, h, h, 0, qscale, 1);
    cvtT(Wkg + wof, h, h); gemm(Xb, bkg + bof, nullptr, Kgb, BS, h, h, 0, 1.0f,   1);
    cvtT(Wvg + wof, h, h); gemm(Xb, bvg + bof, nullptr, Vgb, BS, h, h, 0, 1.0f,   1);
    // attention
    attn_band_kernel<<<dim3(16, B * NHEADS), dim3(256), 0, stream>>>(Qb, Kb, Vb, am, O, S);
    attn_global_kernel<<<dim3(B * NHEADS), dim3(128), 0, stream>>>(Qgb, Kgb, Vgb, am, O, S);
    merge_heads_kernel<<<dim3((unsigned)(((size_t)BS * h) / 256)), dim3(256), 0, stream>>>(O, Ob, S);
    // output projection + residual LN
    cvtT(Wo + wof, h, h);
    gemm(Ob, bo + bof, Y, nullptr, BS, h, h, 0, 1.0f, 0);
    add_ln_kernel<<<dim3(BS), dim3(256), 0, stream>>>(X, Y, ln1g + bof, ln1b + bof, Xb);
    // FFN
    cvtT(W1 + (size_t)l * h * dff, h, dff);
    gemm(Xb, b1 + (size_t)l * dff, nullptr, Yb, BS, dff, h, 1, 1.0f, 0);
    cvtT(W2 + (size_t)l * dff * h, dff, h);
    gemm(Yb, b2 + bof, Y, nullptr, BS, h, dff, 0, 1.0f, 0);
    add_ln_kernel<<<dim3(BS), dim3(256), 0, stream>>>(X, Y, ln2g + bof, ln2b + bof, Xb);
  }

  // pair-classifier head
  pair_gather_kernel<<<dim3(B * P), dim3(256), 0, stream>>>(X, pidx, PEb, S);
  cvtT(F1, 1536, 2304); gemm(PEb, f1, nullptr, Z1b, 256, 2304, 1536, 2, 1.0f, 0);
  cvtT(F2, 2304, 1536); gemm(Z1b, f2, nullptr, Z2b, 256, 1536, 2304, 2, 1.0f, 0);
  cvtT(F3, 1536, 768);  gemm(Z2b, f3, Z3, nullptr, 256, 768, 1536, 2, 1.0f, 0);
  final_kernel<<<dim3(3), dim3(256), 0, stream>>>(Z3, F4, f4, (float*)d_out);
}
